// AttentionRNNLM_47047071761169
// MI455X (gfx1250) — matrix operand order repaired and checked by reading, not compiled, not run on hardware
//
#include <hip/hip_runtime.h>

typedef unsigned short u16;
typedef __attribute__((ext_vector_type(16))) __bf16 v16bf;
typedef __attribute__((ext_vector_type(8)))  float  v8f;
typedef __attribute__((ext_vector_type(4)))  unsigned int u32x4;
typedef __attribute__((ext_vector_type(2)))  float  f32x2;
typedef __attribute__((ext_vector_type(4)))  unsigned int uv4;
typedef __attribute__((ext_vector_type(8)))  int iv8;
typedef __attribute__((ext_vector_type(4)))  int iv4;

union Frag { v16bf v; u32x4 q[2]; };

__device__ __forceinline__ u16 f2bf(float x) {
  unsigned int u = __float_as_uint(x);
  unsigned int r = u + 0x7FFFu + ((u >> 16) & 1u);  // RNE
  return (u16)(r >> 16);
}

// Use the Tensor Data Mover when the toolchain exposes it (device pass only).
#if defined(__HIP_DEVICE_COMPILE__) && __has_builtin(__builtin_amdgcn_tensor_load_to_lds) && \
    __has_builtin(__builtin_amdgcn_s_wait_tensorcnt)
#define USE_TDM 1
#else
#define USE_TDM 0
#endif

#define LDSTR 40  // u16 row stride in LDS (32 data + 8 pad) = 80B (16B aligned)

#if USE_TDM
// One TDM descriptor moves a 128x32 bf16 tile (row stride ld_elems) into LDS,
// inserting 4 DWORDs of padding every 16 DWORDs -> LDS row stride = 40 u16.
__device__ __forceinline__ void tdm_load_tile(const u16* gsrc, u16* lds_dst, int ld_elems)
{
  const unsigned long long ga = (unsigned long long)(uintptr_t)gsrc;
  const unsigned lds_off = (unsigned)(uintptr_t)lds_dst;  // low 32 bits = LDS byte offset
  uv4 g0 = {1u,                                   // count=1 (valid user descriptor)
            lds_off,
            (unsigned)ga,
            (unsigned)(((ga >> 32) & 0x1FFFFFFull) | (2u << 30))};  // type=2 ("image")
  iv8 g1;
  g1[0] = (int)((1u << 16)            // data_size = 1 -> 2 bytes
              | (1u << 20)            // pad_enable
              | (3u << 22)            // pad_interval: every 16 DWORDs
              | (3u << 25));          // pad_amount: 4 DWORDs
  g1[1] = (int)(32u << 16);           // tensor_dim0 = 32 (tile always in-bounds)
  g1[2] = (int)(128u << 16);          // tensor_dim1 = 128
  g1[3] = (int)(32u << 16);           // tile_dim0 = 32
  g1[4] = (int)(128u);                // tile_dim1 = 128, tile_dim2 = 0
  g1[5] = (int)(unsigned)ld_elems;    // tensor_dim0_stride lo (elements)
  g1[6] = 0;                          // stride hi, dim1 stride unused (2D)
  g1[7] = 0;
  iv4 z4 = {0, 0, 0, 0};
#if __clang_major__ >= 23
  iv8 z8 = {0, 0, 0, 0, 0, 0, 0, 0};
  __builtin_amdgcn_tensor_load_to_lds(g0, g1, z4, z4, z8, 0);
#else
  __builtin_amdgcn_tensor_load_to_lds(g0, g1, z4, z4, 0);
#endif
}
#endif

// ---------------------------------------------------------------------------
// Generic bf16 WMMA GEMM:  C[M,N] = alpha * (A[M,K] @ B[N,K]^T) + bias[N]
// A, B bf16 row-major; C fp32 and/or Cb bf16 outputs (nullable, NT stores).
// M,N % 128 == 0, K % 32 == 0. Grid: (N/128, M/128, batch), 256 threads.
// Tiles staged into LDS by the Tensor Data Mover, double-buffered so the DMA
// of tile k+1 overlaps WMMA on tile k (TENSORcnt pipelining).
// causal!=0: skip blocks entirely above the diagonal (never read by softmax).
// ---------------------------------------------------------------------------
__global__ __launch_bounds__(256) void wmma_gemm_bf16(
    const u16* __restrict__ A, const u16* __restrict__ B,
    const float* __restrict__ bias, float* __restrict__ C, u16* __restrict__ Cb,
    int K, int lda, int ldb, int ldC, float alpha,
    long long sA, long long sB, long long sC, int causal)
{
  const int rowBase = blockIdx.y * 128;
  const int colBase = blockIdx.x * 128;
  if (causal && colBase >= rowBase + 128) return;  // fully-masked tile (uniform exit)

  const int tid  = threadIdx.x;
  const int lane = tid & 31;
  const int wave = tid >> 5;
  const int wm   = wave & 3;   // 0..3 -> M position
  const int wn   = wave >> 2;  // 0..1 -> N position

  A += (size_t)blockIdx.z * (size_t)sA;
  B += (size_t)blockIdx.z * (size_t)sB;
  const size_t cOff = (size_t)blockIdx.z * (size_t)sC;

  __shared__ alignas(16) u16 As[2][128 * LDSTR];
  __shared__ alignas(16) u16 Bs[2][128 * LDSTR];

  v8f acc[2][4];
#pragma unroll
  for (int mt = 0; mt < 2; ++mt)
#pragma unroll
    for (int nt = 0; nt < 4; ++nt) acc[mt][nt] = (v8f){};

  const int mrow  = lane & 15;
  const int abase = (lane >= 16) ? 16 : 0;   // A: k-halves 0-15 vs 16-31
  const int bbase = (lane >= 16) ? 16 : 0;   // B: k-halves 0-15 vs 16-31

  int cur = 0;
#if USE_TDM
  if (wave == 0) {
    tdm_load_tile(A + (size_t)rowBase * lda, &As[0][0], lda);
    tdm_load_tile(B + (size_t)colBase * ldb, &Bs[0][0], ldb);
  }
#else
  const int lr = tid >> 1;          // cooperative copy: 128 rows x 32 u16
  const int lc = (tid & 1) << 4;
#endif

  for (int kb = 0; kb < K; kb += 32) {
#if USE_TDM
    if (wave == 0) {
      if (kb + 32 < K) {
        tdm_load_tile(A + (size_t)rowBase * lda + kb + 32, &As[cur ^ 1][0], lda);
        tdm_load_tile(B + (size_t)colBase * ldb + kb + 32, &Bs[cur ^ 1][0], ldb);
        __builtin_amdgcn_s_wait_tensorcnt(2);  // oldest pair (tile kb) complete
      } else {
        __builtin_amdgcn_s_wait_tensorcnt(0);
      }
    }
    __syncthreads();
#else
    {
      const u32x4* ga = (const u32x4*)(A + (size_t)(rowBase + lr) * lda + kb + lc);
      const u32x4* gb = (const u32x4*)(B + (size_t)(colBase + lr) * ldb + kb + lc);
      u32x4* da = (u32x4*)&As[cur][lr * LDSTR + lc];
      u32x4* db = (u32x4*)&Bs[cur][lr * LDSTR + lc];
      da[0] = ga[0]; da[1] = ga[1];
      db[0] = gb[0]; db[1] = gb[1];
    }
    __syncthreads();
#endif

    Frag a[2];
#pragma unroll
    for (int mt = 0; mt < 2; ++mt) {
      const u16* pa = &As[cur][(wm * 32 + mt * 16 + mrow) * LDSTR];
      a[mt].q[0] = *(const u32x4*)(pa + abase);
      a[mt].q[1] = *(const u32x4*)(pa + 8 + abase);
    }
#pragma unroll
    for (int nt = 0; nt < 4; ++nt) {
      const u16* pb = &Bs[cur][(wn * 64 + nt * 16 + mrow) * LDSTR + bbase];
      Frag b;
      b.q[0] = ((const u32x4*)pb)[0];
      b.q[1] = ((const u32x4*)pb)[1];
#pragma unroll
      for (int mt = 0; mt < 2; ++mt)
        acc[mt][nt] = __builtin_amdgcn_wmma_f32_16x16x32_bf16(
            false, a[mt].v, false, b.v, (short)0, acc[mt][nt], false, false);
    }
    __syncthreads();
#if USE_TDM
    cur ^= 1;
#endif
  }

  const int mo = (lane >= 16) ? 8 : 0;
#pragma unroll
  for (int mt = 0; mt < 2; ++mt) {
#pragma unroll
    for (int nt = 0; nt < 4; ++nt) {
      const int col = colBase + wn * 64 + nt * 16 + mrow;
      const float bv = bias ? bias[col] : 0.0f;
#pragma unroll
      for (int r = 0; r < 8; ++r) {
        const int row = rowBase + wm * 32 + mt * 16 + r + mo;
        const float val = acc[mt][nt][r] * alpha + bv;
        const size_t o = cOff + (size_t)row * ldC + col;
        if (C)  __builtin_nontemporal_store(val, &C[o]);       // 1GB logits: NT
        if (Cb) __builtin_nontemporal_store(f2bf(val), &Cb[o]);
      }
    }
  }
}

// ---------------------------------------------------------------------------
__global__ void embed_kernel(const int* __restrict__ x, const float* __restrict__ emb,
                             u16* __restrict__ z, int ntok, int E)
{
  size_t idx = (size_t)blockIdx.x * blockDim.x + threadIdx.x;
  size_t n = (size_t)ntok * E;
  if (idx >= n) return;
  int tok = (int)(idx / E);
  int e   = (int)(idx % E);
  z[idx] = f2bf(emb[(size_t)x[tok] * E + e]);
}

__global__ void cast_bf16_kernel(const float* __restrict__ in, u16* __restrict__ out, size_t n)
{
  size_t idx = (size_t)blockIdx.x * blockDim.x + threadIdx.x;
  if (idx < n) out[idx] = f2bf(in[idx]);
}

// W_hh [1536,512] -> k-pair-major float2 layout: Wp[k2][g] = {W[g][2k2], W[g][2k2+1]}
__global__ void pack_whh_kernel(const float* __restrict__ W, float* __restrict__ Wp)
{
  size_t idx = (size_t)blockIdx.x * blockDim.x + threadIdx.x;
  if (idx >= (size_t)1536 * 512) return;
  int g = (int)(idx >> 9), k = (int)(idx & 511);
  Wp[(((size_t)(k >> 1) * 1536) + g) * 2 + (k & 1)] = W[idx];
}

// in [rows][cols] -> out [cols][rows]; batch via blockIdx.z; block (32,8)
template <typename T>
__global__ void transpose_kernel(const T* __restrict__ in, T* __restrict__ out,
                                 int rows, int cols, long long sIn, long long sOut)
{
  __shared__ T tile[32][33];
  const T* inb = in + (size_t)blockIdx.z * (size_t)sIn;
  T* outb = out + (size_t)blockIdx.z * (size_t)sOut;
  const int c0 = blockIdx.x * 32, r0 = blockIdx.y * 32;
  const int tx = threadIdx.x, ty = threadIdx.y;
#pragma unroll
  for (int i = 0; i < 32; i += 8) {
    int r = r0 + ty + i, c = c0 + tx;
    if (r < rows && c < cols) tile[ty + i][tx] = inb[(size_t)r * cols + c];
  }
  __syncthreads();
#pragma unroll
  for (int i = 0; i < 32; i += 8) {
    int r = c0 + ty + i, c = r0 + tx;
    if (r < cols && c < rows) outb[(size_t)r * rows + c] = tile[tx][ty + i];
  }
}

// ---------------------------------------------------------------------------
// GRU scan: one workgroup per batch element (independent recurrences), 512
// threads = one per hidden column; each thread accumulates all three gates
// with packed v_pk_fma_f32 (2 MAC/lane/clk) over a coalesced float2 stream.
// ---------------------------------------------------------------------------
__global__ __launch_bounds__(512) void gru_kernel(
    const float* __restrict__ xg, const float* __restrict__ Wp,
    const float* __restrict__ b_hh, u16* __restrict__ hs_b)
{
  const int b = blockIdx.x;
  const int j = threadIdx.x;  // hidden column 0..511
  __shared__ alignas(16) float h[512];
  h[j] = 0.0f;
  __syncthreads();

  const f32x2* wp = (const f32x2*)Wp;  // [256][1536] float2 (k-pairs)
  const float bhr = b_hh[j], bhz = b_hh[512 + j], bhn = b_hh[1024 + j];

  for (int t = 0; t < 2048; ++t) {
    f32x2 ar = {0.0f, 0.0f}, az = {0.0f, 0.0f}, an = {0.0f, 0.0f};
#pragma unroll 4
    for (int k2 = 0; k2 < 256; ++k2) {
      const f32x2 h2 = *(const f32x2*)&h[k2 * 2];
      const f32x2 w0 = wp[(size_t)k2 * 1536 + j];
      const f32x2 w1 = wp[(size_t)k2 * 1536 + 512 + j];
      const f32x2 w2 = wp[(size_t)k2 * 1536 + 1024 + j];
      asm("v_pk_fma_f32 %0, %1, %2, %0" : "+v"(ar) : "v"(h2), "v"(w0));
      asm("v_pk_fma_f32 %0, %1, %2, %0" : "+v"(az) : "v"(h2), "v"(w1));
      asm("v_pk_fma_f32 %0, %1, %2, %0" : "+v"(an) : "v"(h2), "v"(w2));
    }
    const float hr = ar[0] + ar[1] + bhr;
    const float hz = az[0] + az[1] + bhz;
    const float hn = an[0] + an[1] + bhn;
    const float* xrow = xg + ((size_t)b * 2048 + t) * 1536;
    const float r = 1.0f / (1.0f + __expf(-(xrow[j] + hr)));
    const float z = 1.0f / (1.0f + __expf(-(xrow[512 + j] + hz)));
    const float n = tanhf(xrow[1024 + j] + r * hn);
    const float hold = h[j];
    __syncthreads();  // all reads of old h done
    const float hnew = (1.0f - z) * n + z * hold;
    h[j] = hnew;
    hs_b[((size_t)b * 2048 + t) * 512 + j] = f2bf(hnew);
    __syncthreads();  // h update visible before next step
  }
}

// ---------------------------------------------------------------------------
// Causal mask + softmax, one block (256 thr) per (b, i) row; bf16 probs out.
// ---------------------------------------------------------------------------
__global__ __launch_bounds__(256) void softmax_causal(
    const float* __restrict__ S, u16* __restrict__ P, int T)
{
  const int row = blockIdx.x;  // b*T + i
  const int i = row % T;
  const int len = i + 1;
  const int tid = threadIdx.x;
  const float* s = S + (size_t)row * T;
  u16* p = P + (size_t)row * T;

  __shared__ float red[256];

  float m = -1e30f;
  for (int j = tid; j < len; j += 256) m = fmaxf(m, s[j]);
  red[tid] = m; __syncthreads();
  for (int st = 128; st > 0; st >>= 1) {
    if (tid < st) red[tid] = fmaxf(red[tid], red[tid + st]);
    __syncthreads();
  }
  m = red[0]; __syncthreads();

  float sum = 0.0f;
  for (int j = tid; j < len; j += 256) sum += __expf(s[j] - m);
  red[tid] = sum; __syncthreads();
  for (int st = 128; st > 0; st >>= 1) {
    if (tid < st) red[tid] += red[tid + st];
    __syncthreads();
  }
  const float inv = 1.0f / red[0];

  for (int j = tid; j < T; j += 256)
    p[j] = (j < len) ? f2bf(__expf(s[j] - m) * inv) : (u16)0;
}

// ---------------------------------------------------------------------------
extern "C" void kernel_launch(void* const* d_in, const int* in_sizes, int n_in,
                              void* d_out, int out_size, void* d_ws, size_t ws_size,
                              hipStream_t stream) {
  (void)in_sizes; (void)n_in; (void)out_size; (void)ws_size;
  const int*   x    = (const int*)  d_in[0];
  const float* emb  = (const float*)d_in[1];
  const float* W_ih = (const float*)d_in[2];
  const float* W_hh = (const float*)d_in[3];
  const float* b_ih = (const float*)d_in[4];
  const float* b_hh = (const float*)d_in[5];
  const float* Wq   = (const float*)d_in[6];
  const float* bq   = (const float*)d_in[7];
  const float* Wk   = (const float*)d_in[8];
  const float* bk   = (const float*)d_in[9];
  const float* Wv   = (const float*)d_in[10];
  const float* bv   = (const float*)d_in[11];
  const float* Wfc  = (const float*)d_in[12];
  const float* bfc  = (const float*)d_in[13];
  float* out = (float*)d_out;

  constexpr int Bn = 4, T = 2048, E = 512, H = 512, V = 32000;
  constexpr int BT = Bn * T, G = 3 * H;

  char* ws = (char*)d_ws;
  size_t off = 0;
  auto take = [&](size_t bytes) -> void* {
    void* p = ws + off;
    off += (bytes + 255) & ~(size_t)255;
    return p;
  };

  u16*   z_b    = (u16*)  take((size_t)BT * E * 2);
  u16*   Wih_b  = (u16*)  take((size_t)G * E * 2);
  float* Wp     = (float*)take((size_t)H * G * 4);
  u16*   Wq_b   = (u16*)  take((size_t)H * H * 2);
  u16*   Wk_b   = (u16*)  take((size_t)H * H * 2);
  u16*   Wv_b   = (u16*)  take((size_t)H * H * 2);
  u16*   Wfc_b  = (u16*)  take((size_t)V * H * 2);
  float* xg     = (float*)take((size_t)BT * G * 4);
  u16*   hs_b   = (u16*)  take((size_t)BT * H * 2);
  u16*   q_b    = (u16*)  take((size_t)BT * H * 2);
  u16*   k_b    = (u16*)  take((size_t)BT * H * 2);
  u16*   v_b    = (u16*)  take((size_t)BT * H * 2);
  u16*   vT_b   = (u16*)  take((size_t)BT * H * 2);
  float* S      = (float*)take((size_t)Bn * T * T * 4);
  u16*   attn_b = (u16*)  take((size_t)Bn * T * T * 2);
  u16*   ctx_b  = (u16*)  take((size_t)BT * H * 2);

  auto castb = [&](const float* src, u16* dst, size_t n) {
    cast_bf16_kernel<<<(unsigned)((n + 255) / 256), 256, 0, stream>>>(src, dst, n);
  };
  auto gemm = [&](const u16* A, const u16* Bm, const float* bias, float* C, u16* Cb,
                  int M, int N, int K, int lda, int ldb, int ldC, float alpha,
                  long long sA, long long sB, long long sC, int batch, int causal) {
    dim3 grid(N / 128, M / 128, batch);
    wmma_gemm_bf16<<<grid, 256, 0, stream>>>(A, Bm, bias, C, Cb, K, lda, ldb, ldC,
                                             alpha, sA, sB, sC, causal);
  };

  // weight preparation
  castb(W_ih, Wih_b, (size_t)G * E);
  castb(Wq, Wq_b, (size_t)H * H);
  castb(Wk, Wk_b, (size_t)H * H);
  castb(Wv, Wv_b, (size_t)H * H);
  castb(Wfc, Wfc_b, (size_t)V * H);
  pack_whh_kernel<<<(G * E + 255) / 256, 256, 0, stream>>>(W_hh, Wp);

  // embedding gather
  {
    size_t n = (size_t)BT * E;
    embed_kernel<<<(unsigned)((n + 255) / 256), 256, 0, stream>>>(x, emb, z_b, BT, E);
  }

  // xg = z @ W_ih^T + b_ih  (fp32 out for the recurrence)
  gemm(z_b, Wih_b, b_ih, xg, nullptr, BT, G, E, E, E, G, 1.0f, 0, 0, 0, 1, 0);

  // sequential GRU scan: one WGP per batch element
  gru_kernel<<<Bn, 512, 0, stream>>>(xg, Wp, b_hh, hs_b);

  // q,k,v projections (bf16 outputs)
  gemm(hs_b, Wq_b, bq, nullptr, q_b, BT, H, H, H, H, H, 1.0f, 0, 0, 0, 1, 0);
  gemm(hs_b, Wk_b, bk, nullptr, k_b, BT, H, H, H, H, H, 1.0f, 0, 0, 0, 1, 0);
  gemm(hs_b, Wv_b, bv, nullptr, v_b, BT, H, H, H, H, H, 1.0f, 0, 0, 0, 1, 0);

  // v^T per batch for attn@v (B-operand must be [N,K] row-major)
  transpose_kernel<u16><<<dim3(H / 32, T / 32, Bn), dim3(32, 8), 0, stream>>>(
      v_b, vT_b, T, H, (long long)T * H, (long long)H * T);

  // scores = (q @ k^T) / sqrt(H)  (batched, upper-triangle tiles skipped)
  gemm(q_b, k_b, nullptr, S, nullptr, T, T, H, H, H, T, 0.044194173824159216f,
       (long long)T * H, (long long)T * H, (long long)T * T, Bn, 1);

  // causal softmax -> bf16 probabilities
  softmax_causal<<<BT, 256, 0, stream>>>(S, attn_b, T);

  // ctx = attn @ v   (batched)
  gemm(attn_b, vT_b, nullptr, nullptr, ctx_b, T, H, T, T, T, H, 1.0f,
       (long long)T * T, (long long)H * T, (long long)T * H, Bn, 0);

  // logits = ctx @ Wfc^T + bfc  -> d_out [B,T,V] fp32
  gemm(ctx_b, Wfc_b, bfc, out, nullptr, BT, V, H, H, H, V, 1.0f, 0, 0, 0, 1, 0);
}
